// LocalSelfAttention_8813272891488
// MI455X (gfx1250) — matrix operand order repaired and checked by reading, not compiled, not run on hardware
//
#include <hip/hip_runtime.h>
#include <hip/hip_bf16.h>

// ---------------------------------------------------------------------------
// Problem constants (match reference)
// ---------------------------------------------------------------------------
#define B_   2
#define P_   4096
#define DIM_ 256
#define H_   8
#define KNN_ 32
#define DH_  32
#define SCALE_ 0.17677669529663687f   // 32^-0.5
#define KSTEPS 8                      // DIM_ / 32

typedef __attribute__((ext_vector_type(16))) __bf16       v16bf;
typedef __attribute__((ext_vector_type(8)))  float        v8f;
typedef __attribute__((ext_vector_type(4)))  unsigned int u32x4;

union ABFrag { u32x4 q[2]; v16bf v; };

#if __has_builtin(__builtin_amdgcn_sched_barrier)
#define SCHED_FENCE() __builtin_amdgcn_sched_barrier(0)
#else
#define SCHED_FENCE()
#endif

__device__ __forceinline__ __bf16 f2bf(float f) {
    unsigned u = __builtin_bit_cast(unsigned, f);
    unsigned r = (u + 0x7FFFu + ((u >> 16) & 1u)) >> 16;
    unsigned short s = (unsigned short)r;
    return __builtin_bit_cast(__bf16, s);
}

// gfx1250 async copy: global -> LDS, 16B per lane, tracked by ASYNCcnt.
__device__ __forceinline__ void async_g2l_b128(unsigned lds_addr, unsigned goff,
                                               unsigned long long gbase) {
    asm volatile("global_load_async_to_lds_b128 %0, %1, %2"
                 :: "v"(lds_addr), "v"(goff), "s"(gbase) : "memory");
}
__device__ __forceinline__ void wait_asynccnt0() {
    asm volatile("s_wait_asynccnt 0" ::: "memory");
}

// ---------------------------------------------------------------------------
// Shared-memory layout for the fused attention kernel (dynamic LDS)
// ---------------------------------------------------------------------------
#define H_STRIDE  (DIM_ + 8)    // bf16 elements per h row (16B-aligned stride)
#define KV_STRIDE (DIM_ + 4)    // f32 elements per K/V row (1040B: breaks bank conflicts)

constexpr size_t OFF_H   = 0;
constexpr size_t OFF_PB  = (OFF_H  + (size_t)KNN_ * H_STRIDE * 2  + 127) & ~(size_t)127;
constexpr size_t OFF_K   = (OFF_PB + (size_t)KNN_ * DIM_      * 4 + 127) & ~(size_t)127;
constexpr size_t OFF_V   = (OFF_K  + (size_t)KNN_ * KV_STRIDE * 4 + 127) & ~(size_t)127;
constexpr size_t OFF_ATT = (OFF_V  + (size_t)KNN_ * KV_STRIDE * 4 + 127) & ~(size_t)127;
constexpr size_t OFF_IDX = OFF_ATT + (size_t)KNN_ * H_ * 4;
constexpr size_t OFF_RP  = OFF_IDX + (size_t)KNN_ * 4;
constexpr size_t SMEM_SZ = OFF_RP  + (size_t)KNN_ * 3 * 4;      // ~118 KB

// ---------------------------------------------------------------------------
// f32 -> bf16 conversion
// ---------------------------------------------------------------------------
__global__ void f32_to_bf16(const float* __restrict__ in, __bf16* __restrict__ out, int n) {
    int i = blockIdx.x * blockDim.x + threadIdx.x;
    if (i < n) out[i] = f2bf(in[i]);
}

// ---------------------------------------------------------------------------
// Brute-force kNN (K=32) with LDS candidate tiling + register insertion sort.
// ---------------------------------------------------------------------------
__global__ __launch_bounds__(256)
void knn_kernel(const float* __restrict__ xyz,
                int* __restrict__ idx_out, float* __restrict__ relpos_out) {
    const int b  = blockIdx.y;
    const int qp = blockIdx.x * 256 + threadIdx.x;

    __shared__ float sx[256], sy[256], sz[256];

    const float qx = xyz[((size_t)b * P_ + qp) * 3 + 0];
    const float qy = xyz[((size_t)b * P_ + qp) * 3 + 1];
    const float qz = xyz[((size_t)b * P_ + qp) * 3 + 2];

    float d[KNN_];
    int   id[KNN_];
#pragma unroll
    for (int t = 0; t < KNN_; ++t) { d[t] = 3.4e38f; id[t] = 0; }

    for (int tile = 0; tile < P_; tile += 256) {
        const int j = tile + threadIdx.x;
        sx[threadIdx.x] = xyz[((size_t)b * P_ + j) * 3 + 0];
        sy[threadIdx.x] = xyz[((size_t)b * P_ + j) * 3 + 1];
        sz[threadIdx.x] = xyz[((size_t)b * P_ + j) * 3 + 2];
        __syncthreads();
        for (int u = 0; u < 256; ++u) {
            const float dx = sx[u] - qx, dy = sy[u] - qy, dz = sz[u] - qz;
            const float dist = fmaf(dx, dx, fmaf(dy, dy, dz * dz));
            if (dist < d[KNN_ - 1]) {
                d[KNN_ - 1] = dist; id[KNN_ - 1] = tile + u;
#pragma unroll
                for (int t = KNN_ - 2; t >= 0; --t) {
                    if (d[t + 1] < d[t]) {
                        float td = d[t]; d[t] = d[t + 1]; d[t + 1] = td;
                        int   ti = id[t]; id[t] = id[t + 1]; id[t + 1] = ti;
                    }
                }
            }
        }
        __syncthreads();
    }

    const size_t base = ((size_t)b * P_ + qp) * KNN_;
#pragma unroll 4
    for (int t = 0; t < KNN_; ++t) {
        idx_out[base + t] = id[t];
        const size_t nb = ((size_t)b * P_ + id[t]) * 3;
        relpos_out[(base + t) * 3 + 0] = xyz[nb + 0] - qx;
        relpos_out[(base + t) * 3 + 1] = xyz[nb + 1] - qy;
        relpos_out[(base + t) * 3 + 2] = xyz[nb + 2] - qz;
    }
}

// ---------------------------------------------------------------------------
// bf16 WMMA GEMM:  C(M,N) f32 = A(M,DIM_) @ W(N,DIM_)^T  (+ optional bias[n])
// Block = 256 threads = 8 waves; wave tile = 16(M) x 64(N); WG tile 64 x 128.
// K loop fully unrolled (8 steps of 32); next-step fragments are loaded and
// then pinned above the current WMMA group with a sched_barrier so the
// machine scheduler cannot sink them back onto their consumers.
// ---------------------------------------------------------------------------
__global__ __launch_bounds__(256)
void gemm_bf16_nt(const __bf16* __restrict__ A, const __bf16* __restrict__ Bw,
                  const float* __restrict__ bias, float* __restrict__ C,
                  int M, int N) {
    const int lane = threadIdx.x & 31;
    const int wave = threadIdx.x >> 5;
    const int row0 = blockIdx.x * 64  + (wave & 3) * 16;
    const int nb   = blockIdx.y * 128 + (wave >> 2) * 64;

    v8f acc[4] = {};

    const int mA = lane & 15;           // A row inside tile
    const int kA = (lane >> 4) * 16;    // A K sub-offset (0 or 16)
    const int nB = lane & 15;           // B column inside tile
    const int kB = (lane >> 4) * 16;    // B K sub-offset (0 or 16)

    const __bf16* arow = A + (size_t)(row0 + mA) * DIM_;

    auto loadA = [&](ABFrag& a, int kk) {
        const __bf16* ap = arow + kk + kA;
        a.q[0] = *(const u32x4*)(ap);
        a.q[1] = *(const u32x4*)(ap + 8);
    };
    auto loadB = [&](ABFrag* bb, int kk) {
#pragma unroll
        for (int t = 0; t < 4; ++t) {
            const __bf16* bp = Bw + (size_t)(nb + t * 16 + nB) * DIM_ + kk + kB;
            bb[t].q[0] = *(const u32x4*)(bp);
            bb[t].q[1] = *(const u32x4*)(bp + 8);
        }
    };

    ABFrag a[2];
    ABFrag bfr[2][4];
    loadA(a[0], 0);
    loadB(bfr[0], 0);
    __builtin_prefetch(arow + 128, 0, 1);   // gfx1250 global_prefetch on the A stream

#pragma unroll
    for (int s = 0; s < KSTEPS; ++s) {
        const int cur = s & 1, nxt = cur ^ 1;
        if (s + 1 < KSTEPS) {               // issue next k-step's fragment loads
            loadA(a[nxt], (s + 1) * 32);
            loadB(bfr[nxt], (s + 1) * 32);
        }
        SCHED_FENCE();                      // loads stay above the WMMA group
#pragma unroll
        for (int t = 0; t < 4; ++t)
            acc[t] = __builtin_amdgcn_wmma_f32_16x16x32_bf16(
                false, a[cur].v, false, bfr[cur][t].v, (short)0, acc[t], false, false);
        SCHED_FENCE();                      // WMMAs stay above next step's loads
    }

    const int mHi = (lane >> 4) * 8;
#pragma unroll
    for (int t = 0; t < 4; ++t) {
        const int n = nb + t * 16 + nB;
        const float bv = bias ? bias[n] : 0.0f;
#pragma unroll
        for (int r = 0; r < 8; ++r) {
            const int m = row0 + mHi + r;
            C[(size_t)m * N + n] = acc[t][r] + bv;
        }
    }
}

// ---------------------------------------------------------------------------
// Fused per-point kernel:
//   - async-gather 32 neighbor K/V rows into LDS (ASYNCcnt) while...
//   - pos-MLP layer1 (VALU) and layer2 (WMMA 32x256x256) execute,
//   - then per-head scores, wave32 softmax over K, weighted sum.
// One workgroup (256 threads = 8 waves) per (batch, point). Dynamic LDS.
// ---------------------------------------------------------------------------
__global__ __launch_bounds__(256)
void attn_kernel(const float* __restrict__ qm, const float* __restrict__ km,
                 const float* __restrict__ vm,
                 const int* __restrict__ idx, const float* __restrict__ relpos,
                 const float* __restrict__ W1, const float* __restrict__ b1,
                 const __bf16* __restrict__ W2bf, const float* __restrict__ b2,
                 __bf16* __restrict__ out_bf) {
    extern __shared__ char smem[];
    __bf16 (*h_s)[H_STRIDE] = (__bf16(*)[H_STRIDE])(smem + OFF_H);
    float  (*pb_s)[DIM_]    = (float (*)[DIM_])   (smem + OFF_PB);
    float*  k_s             = (float*)            (smem + OFF_K);
    float*  v_s             = (float*)            (smem + OFF_V);
    float  (*attn_s)[H_]    = (float (*)[H_])     (smem + OFF_ATT);
    int*    idx_s           = (int*)              (smem + OFF_IDX);
    float  (*rp_s)[3]       = (float (*)[3])      (smem + OFF_RP);

    const int b = blockIdx.y, p = blockIdx.x;
    const int tid = threadIdx.x, lane = tid & 31, wave = tid >> 5;
    const size_t pbase = (size_t)b * P_ + p;

    if (tid < KNN_) {
        idx_s[tid] = idx[pbase * KNN_ + tid];
        rp_s[tid][0] = relpos[(pbase * KNN_ + tid) * 3 + 0];
        rp_s[tid][1] = relpos[(pbase * KNN_ + tid) * 3 + 1];
        rp_s[tid][2] = relpos[(pbase * KNN_ + tid) * 3 + 2];
    }
    __syncthreads();

    // ---- kick off async gather of the 32 neighbor K/V rows (64 KB) --------
    {
        const unsigned kl = (unsigned)(size_t)k_s;
        const unsigned vl = (unsigned)(size_t)v_s;
        const unsigned long long kg =
            (unsigned long long)(size_t)(km + (size_t)b * P_ * DIM_);
        const unsigned long long vg =
            (unsigned long long)(size_t)(vm + (size_t)b * P_ * DIM_);
#pragma unroll
        for (int i = 0; i < 8; ++i) {
            const int t     = tid + i * 256;        // 0..2047 transfers of 16B
            const int row   = t >> 6;               // neighbor 0..31
            const int chunk = (t & 63) << 4;        // byte offset within row
            const unsigned go = (unsigned)idx_s[row] * (DIM_ * 4u) + (unsigned)chunk;
            async_g2l_b128(kl + row * (KV_STRIDE * 4u) + chunk, go, kg);
            async_g2l_b128(vl + row * (KV_STRIDE * 4u) + chunk, go, vg);
        }
    }

    // ---- layer 1: h[k][c] = relu(relpos[k] . W1[c] + b1[c]), c = tid ------
    {
        const float w0 = W1[tid * 3 + 0], w1 = W1[tid * 3 + 1], w2 = W1[tid * 3 + 2];
        const float bb = b1[tid];
#pragma unroll 8
        for (int k = 0; k < KNN_; ++k) {
            float v = fmaf(rp_s[k][0], w0, fmaf(rp_s[k][1], w1, fmaf(rp_s[k][2], w2, bb)));
            h_s[k][tid] = f2bf(v > 0.0f ? v : 0.0f);
        }
    }
    __syncthreads();

    // ---- layer 2: pos_bias(32x256) = h(32x256) @ W2^T, WMMA, pipelined ----
    {
        const int mt  = wave & 1;            // neighbor tile (0..1)
        const int nc  = (wave >> 1) * 64;    // channel chunk
        const int mAr = mt * 16 + (lane & 15);
        const int kA  = (lane >> 4) * 16;
        const int nB  = lane & 15;
        const int kB  = (lane >> 4) * 16;

        auto loadA = [&](ABFrag& aa, int kk) {
            aa.q[0] = *(const u32x4*)&h_s[mAr][kk + kA];
            aa.q[1] = *(const u32x4*)&h_s[mAr][kk + kA + 8];
        };
        auto loadB = [&](ABFrag* bb, int kk) {
#pragma unroll
            for (int t = 0; t < 4; ++t) {
                const __bf16* bp = W2bf + (size_t)(nc + t * 16 + nB) * DIM_ + kk + kB;
                bb[t].q[0] = *(const u32x4*)(bp);
                bb[t].q[1] = *(const u32x4*)(bp + 8);
            }
        };

        v8f acc[4] = {};
        ABFrag a[2];
        ABFrag bfr[2][4];
        loadA(a[0], 0);
        loadB(bfr[0], 0);

#pragma unroll
        for (int s = 0; s < KSTEPS; ++s) {
            const int cur = s & 1, nxt = cur ^ 1;
            if (s + 1 < KSTEPS) {
                loadA(a[nxt], (s + 1) * 32);
                loadB(bfr[nxt], (s + 1) * 32);
            }
            SCHED_FENCE();
#pragma unroll
            for (int t = 0; t < 4; ++t)
                acc[t] = __builtin_amdgcn_wmma_f32_16x16x32_bf16(
                    false, a[cur].v, false, bfr[cur][t].v, (short)0, acc[t], false, false);
            SCHED_FENCE();
        }

        const int mHi = mt * 16 + (lane >> 4) * 8;
#pragma unroll
        for (int t = 0; t < 4; ++t) {
            const int n = nc + t * 16 + nB;
            const float bv = b2[n];
#pragma unroll
            for (int r = 0; r < 8; ++r)
                pb_s[mHi + r][n] = acc[t][r] + bv;
        }
    }

    wait_asynccnt0();      // neighbor K/V rows now resident in LDS
    __syncthreads();

    // ---- scores + softmax: wave = head, lane = neighbor -------------------
    {
        const int h = wave;
        const float* qp_ = qm + pbase * DIM_ + h * DH_;
        const float* krow = k_s + lane * KV_STRIDE + h * DH_;
        float s = 0.0f;
#pragma unroll
        for (int dh = 0; dh < DH_; ++dh)
            s = fmaf(qp_[dh], krow[dh] + pb_s[lane][h * DH_ + dh], s);
        s *= SCALE_;
        float mx = s;
#pragma unroll
        for (int o = 16; o > 0; o >>= 1) mx = fmaxf(mx, __shfl_xor(mx, o, 32));
        const float e = __expf(s - mx);
        float sum = e;
#pragma unroll
        for (int o = 16; o > 0; o >>= 1) sum += __shfl_xor(sum, o, 32);
        attn_s[lane][h] = e / sum;
    }
    __syncthreads();

    // ---- weighted sum over neighbors: c = tid -----------------------------
    {
        const int h = tid >> 5;   // DH_ == 32, so channel/32 == head
        float o = 0.0f;
#pragma unroll 8
        for (int k = 0; k < KNN_; ++k)
            o = fmaf(attn_s[k][h], v_s[k * KV_STRIDE + tid] + pb_s[k][tid], o);
        out_bf[pbase * DIM_ + tid] = f2bf(o);
    }
}

// ---------------------------------------------------------------------------
// Host launcher
// ---------------------------------------------------------------------------
extern "C" void kernel_launch(void* const* d_in, const int* in_sizes, int n_in,
                              void* d_out, int out_size, void* d_ws, size_t ws_size,
                              hipStream_t stream) {
    (void)in_sizes; (void)n_in; (void)out_size; (void)ws_size;

    const float* xyz   = (const float*)d_in[0];
    const float* feats = (const float*)d_in[1];
    const float* Wq    = (const float*)d_in[2];
    const float* Wk    = (const float*)d_in[3];
    const float* Wv    = (const float*)d_in[4];
    const float* Wp    = (const float*)d_in[5];
    const float* bp    = (const float*)d_in[6];
    const float* W1    = (const float*)d_in[7];
    const float* b1    = (const float*)d_in[8];
    const float* W2    = (const float*)d_in[9];
    const float* b2    = (const float*)d_in[10];

    constexpr size_t NPTS   = (size_t)B_ * P_;          // 8192 rows
    constexpr size_t NFEAT  = NPTS * DIM_;              // 2,097,152
    constexpr size_t NW     = (size_t)DIM_ * DIM_;      // 65,536

    char* ws = (char*)d_ws;
    size_t off = 0;
    auto alloc = [&](size_t bytes) { char* pph = ws + off; off += (bytes + 255) & ~(size_t)255; return pph; };

    __bf16* feats_bf = (__bf16*)alloc(NFEAT * 2);
    __bf16* Wq_bf    = (__bf16*)alloc(NW * 2);
    __bf16* Wk_bf    = (__bf16*)alloc(NW * 2);
    __bf16* Wv_bf    = (__bf16*)alloc(NW * 2);
    __bf16* Wp_bf    = (__bf16*)alloc(NW * 2);
    __bf16* W2_bf    = (__bf16*)alloc(NW * 2);
    float*  qmat     = (float*) alloc(NFEAT * 4);
    float*  kmat     = (float*) alloc(NFEAT * 4);
    float*  vmat     = (float*) alloc(NFEAT * 4);
    int*    idx      = (int*)   alloc(NPTS * KNN_ * 4);
    float*  relpos   = (float*) alloc(NPTS * KNN_ * 3 * 4);
    __bf16* attn_bf  = (__bf16*)alloc(NFEAT * 2);

    // 1) conversions to bf16
    f32_to_bf16<<<(NFEAT + 255) / 256, 256, 0, stream>>>(feats, feats_bf, (int)NFEAT);
    f32_to_bf16<<<(NW + 255) / 256, 256, 0, stream>>>(Wq, Wq_bf, (int)NW);
    f32_to_bf16<<<(NW + 255) / 256, 256, 0, stream>>>(Wk, Wk_bf, (int)NW);
    f32_to_bf16<<<(NW + 255) / 256, 256, 0, stream>>>(Wv, Wv_bf, (int)NW);
    f32_to_bf16<<<(NW + 255) / 256, 256, 0, stream>>>(Wp, Wp_bf, (int)NW);
    f32_to_bf16<<<(NW + 255) / 256, 256, 0, stream>>>(W2, W2_bf, (int)NW);

    // 2) kNN + relpos
    knn_kernel<<<dim3(P_ / 256, B_), 256, 0, stream>>>(xyz, idx, relpos);

    // 3) Q/K/V projections (WMMA bf16, pinned software pipeline)
    dim3 ggrid((unsigned)(NPTS / 64), DIM_ / 128);
    gemm_bf16_nt<<<ggrid, 256, 0, stream>>>(feats_bf, Wq_bf, nullptr, qmat, (int)NPTS, DIM_);
    gemm_bf16_nt<<<ggrid, 256, 0, stream>>>(feats_bf, Wk_bf, nullptr, kmat, (int)NPTS, DIM_);
    gemm_bf16_nt<<<ggrid, 256, 0, stream>>>(feats_bf, Wv_bf, nullptr, vmat, (int)NPTS, DIM_);

    // 4) fused pos-MLP + attention (WMMA + async LDS gather)
    attn_kernel<<<dim3(P_, B_), 256, (unsigned)SMEM_SZ, stream>>>(
        qmat, kmat, vmat, idx, relpos, W1, b1, W2_bf, b2, attn_bf);

    // 5) output projection with bias -> d_out (f32)
    gemm_bf16_nt<<<ggrid, 256, 0, stream>>>(attn_bf, Wp_bf, bp, (float*)d_out,
                                            (int)NPTS, DIM_);
}
